// BiDecoderRNN_87205015977991
// MI455X (gfx1250) — matrix operand order repaired and checked by reading, not compiled, not run on hardware
//
#include <hip/hip_runtime.h>
#include <hip/hip_bf16.h>
#include <math.h>

// ---------------------------------------------------------------------------
// BiDecoderRNN forward for MI455X (gfx1250, wave32, WMMA).
// All GEMMs run through v_wmma_f32_16x16x32_bf16 (bf16 A/B, f32 accumulate)
// with 2x2 register tiling: each wave produces a 32x32 output tile (4 WMMA
// per 32-K chunk, each A/B fragment reused twice -> 2x arithmetic intensity).
// Dominant cost: vocab projection [1600,1024]x[1024,32000] (~105 GFLOP).
// Workspace usage ~128 MB (bf16 weight copies + activations).
// ---------------------------------------------------------------------------

typedef __bf16 bf16;
typedef __attribute__((ext_vector_type(16))) __bf16 v16bf;
typedef __attribute__((ext_vector_type(8)))  float  v8f;

static constexpr int kSD = 50;
static constexpr long kPvocabElems = (long)kSD * 32 * 32000;   // 51,200,000

// ---------------------------------------------------------------------------
// f32 -> bf16 conversion (one thread per element)
// ---------------------------------------------------------------------------
__global__ void convert_f32_bf16_kernel(const float* __restrict__ s,
                                        bf16* __restrict__ d, int n) {
  int i = blockIdx.x * 256 + threadIdx.x;
  if (i < n) d[i] = (bf16)s[i];
}

// dst = a + b (combined LSTM biases)
__global__ void addvec_kernel(const float* __restrict__ a, const float* __restrict__ b,
                              float* __restrict__ d, int n) {
  int i = blockIdx.x * 256 + threadIdx.x;
  if (i < n) d[i] = a[i] + b[i];
}

// ---------------------------------------------------------------------------
// reduce_dim: fold bidirectional encoder state and project 2H->H.
// new_enc[l,b,n] = b_red[n] + sum_k concat(hid[l],hid[l+2])[b,k] * W_red[n,k]
// ---------------------------------------------------------------------------
__global__ void reduce_dim_kernel(const float* __restrict__ hid, const float* __restrict__ cel,
                                  const float* __restrict__ W, const float* __restrict__ bias,
                                  float* __restrict__ h0, float* __restrict__ h1,
                                  float* __restrict__ c0, float* __restrict__ c1) {
  int id = blockIdx.x * 256 + threadIdx.x;        // 0 .. 65535
  int which = id >> 14;                           // 0:h0 1:h1 2:c0 3:c1
  int r = id & 16383;
  int b = r >> 9;
  int n = r & 511;
  int l = which & 1;
  const float* src = (which < 2) ? hid : cel;
  const float* s0 = src + ((long)(l       * 32 + b)) * 512;
  const float* s1 = src + ((long)((l + 2) * 32 + b)) * 512;
  const float* w  = W + (long)n * 1024;
  float acc = bias[n];
  for (int k = 0; k < 512; ++k) acc += s0[k] * w[k];
  for (int k = 0; k < 512; ++k) acc += s1[k] * w[512 + k];
  float* dst = (which == 0) ? h0 : (which == 1) ? h1 : (which == 2) ? c0 : c1;
  dst[r] = acc;
}

// ---------------------------------------------------------------------------
// WMMA fragment loaders (layouts per ISA 7.12.2, wave32).
// A (16-bit, 16x32): lanes 0-15 = rows, VGPR j<4: K=2j,2j+1 (+half*8),
//                    VGPR j>=4: K=16+2(j-4) (+half*8).
// B (32x16): lane = output column; same K order as A (K=2j,2j+1 / 16+2(j-4), +half*8).
// Pair loads are dword-aligned; the compiler merges them into b128 loads.
// ---------------------------------------------------------------------------
__device__ __forceinline__ v16bf load_a_frag(const bf16* __restrict__ Ap, int half) {
  v16bf a;
#pragma unroll
  for (int j = 0; j < 8; ++j) {
    int kk = ((j < 4) ? 2 * j : 16 + 2 * (j - 4)) + half * 8;
    union { unsigned int u; bf16 h[2]; } cv;
    cv.u = *reinterpret_cast<const unsigned int*>(Ap + kk);
    a[2 * j]     = cv.h[0];
    a[2 * j + 1] = cv.h[1];
  }
  return a;
}

__device__ __forceinline__ v16bf load_b_frag(const bf16* __restrict__ Bp, int half) {
  v16bf b;
#pragma unroll
  for (int j = 0; j < 8; ++j) {
    int kk = ((j < 4) ? 2 * j : 16 + 2 * (j - 4)) + half * 8;
    union { unsigned int u; bf16 h[2]; } cv;
    cv.u = *reinterpret_cast<const unsigned int*>(Bp + kk);
    b[2 * j]     = cv.h[0];
    b[2 * j + 1] = cv.h[1];
  }
  return b;
}

// ---------------------------------------------------------------------------
// Generic bf16 WMMA GEMM:  C[M,N] = A[M,K] * B[N,K]^T + bias[N]
// A and B may each be split at k == Ksplit between two buffers (used for the
// LSTM's  x@W_ih^T + h@W_hh^T  without concatenation copies).
// 2x2 register tiling: one wave -> 32x32 output tile, 4 accumulators,
// 4 WMMAs per 32-K chunk with A/B fragment reuse.
// Block = (32,4) waves covering 32 rows x 128 cols; grid = (N/128, M/32).
// Optionally also emits a bf16 copy of C (to feed the next GEMM).
// ---------------------------------------------------------------------------
__global__ void gemm_bf16_wmma_kernel(
    const bf16* __restrict__ A1, int lda1, const bf16* __restrict__ A2, int lda2,
    const bf16* __restrict__ B1, int ldb1, const bf16* __restrict__ B2, int ldb2,
    int Ksplit, int K,
    const float* __restrict__ bias,
    float* __restrict__ C, int ldc,
    bf16* __restrict__ Cbf, int ldcbf) {
  const int lane = threadIdx.x;          // 0..31
  const int half = lane >> 4;            // 0/1
  const int mr   = lane & 15;
  const int n0 = (blockIdx.x * blockDim.y + threadIdx.y) * 32;   // wave: 32 cols
  const int m0 = blockIdx.y * 32;                                // block: 32 rows
  const long arow0 = m0 + mr,      arow1 = m0 + 16 + mr;         // A rows (lane)
  const long brow0 = n0 + mr,      brow1 = n0 + 16 + mr;         // W rows (lane)

  v8f acc00, acc01, acc10, acc11;
  const float bv0 = bias ? bias[n0 + mr]      : 0.0f;
  const float bv1 = bias ? bias[n0 + 16 + mr] : 0.0f;
#pragma unroll
  for (int j = 0; j < 8; ++j) {
    acc00[j] = bv0; acc10[j] = bv0;
    acc01[j] = bv1; acc11[j] = bv1;
  }

  for (int k0 = 0; k0 < K; k0 += 32) {
    const bf16* ApBase;
    long la;
    if (k0 < Ksplit) { ApBase = A1 + k0;            la = lda1; }
    else             { ApBase = A2 + (k0 - Ksplit); la = lda2; }
    const bf16* BpBase;
    long lb;
    if (k0 < Ksplit) { BpBase = B1 + k0;            lb = ldb1; }
    else             { BpBase = B2 + (k0 - Ksplit); lb = ldb2; }

    v16bf a0 = load_a_frag(ApBase + arow0 * la, half);
    v16bf a1 = load_a_frag(ApBase + arow1 * la, half);
    v16bf b0 = load_b_frag(BpBase + brow0 * lb, half);
    v16bf b1 = load_b_frag(BpBase + brow1 * lb, half);

    acc00 = __builtin_amdgcn_wmma_f32_16x16x32_bf16(false, a0, false, b0,
                                                    (short)0, acc00, false, false);
    acc01 = __builtin_amdgcn_wmma_f32_16x16x32_bf16(false, a0, false, b1,
                                                    (short)0, acc01, false, false);
    acc10 = __builtin_amdgcn_wmma_f32_16x16x32_bf16(false, a1, false, b0,
                                                    (short)0, acc10, false, false);
    acc11 = __builtin_amdgcn_wmma_f32_16x16x32_bf16(false, a1, false, b1,
                                                    (short)0, acc11, false, false);
  }

  // C/D layout: lane -> n = ncol + (lane&15); VGPR j -> m = mrow + half*8 + j.
#pragma unroll
  for (int j = 0; j < 8; ++j) {
    long mA = m0 + half * 8 + j;
    long mB = m0 + 16 + half * 8 + j;
    long nA = n0 + mr;
    long nB = n0 + 16 + mr;
    C[mA * ldc + nA] = acc00[j];
    C[mA * ldc + nB] = acc01[j];
    C[mB * ldc + nA] = acc10[j];
    C[mB * ldc + nB] = acc11[j];
    if (Cbf) {
      Cbf[mA * ldcbf + nA] = (bf16)acc00[j];
      Cbf[mA * ldcbf + nB] = (bf16)acc01[j];
      Cbf[mB * ldcbf + nA] = (bf16)acc10[j];
      Cbf[mB * ldcbf + nB] = (bf16)acc11[j];
    }
  }
}

// ---------------------------------------------------------------------------
// LSTM elementwise cell update (gate order i,f,g,o along the 4H axis).
// ---------------------------------------------------------------------------
__global__ void lstm_cell_kernel(const float* __restrict__ g, float* __restrict__ c,
                                 float* __restrict__ hf, bf16* __restrict__ hb,
                                 bf16* __restrict__ pv, int pvStride) {
  int id = blockIdx.x * 256 + threadIdx.x;
  if (id >= 32 * 512) return;
  int b = id >> 9, n = id & 511;
  const float* gr = g + (long)b * 2048;
  float gi = gr[n], gf = gr[512 + n], gg = gr[1024 + n], go = gr[1536 + n];
  float si = 1.0f / (1.0f + expf(-gi));
  float sf = 1.0f / (1.0f + expf(-gf));
  float so = 1.0f / (1.0f + expf(-go));
  float cn = sf * c[id] + si * tanhf(gg);
  float hn = so * tanhf(cn);
  c[id]  = cn;
  hf[id] = hn;
  hb[id] = (bf16)hn;
  if (pv) pv[(long)b * pvStride + n] = (bf16)hn;
}

// ---------------------------------------------------------------------------
// Fused additive attention: one block per (t,b) pair (grid=1600, block=256).
//   e[s]   = sum_a v[a] * tanh(enc_part[s,b,a] + dec_part[t,b,a])
//   attn   = softmax_s(e)
//   ctx[e] = sum_s attn[s] * output_enc[s,b,e]   -> bf16 into pv_in[:,512:]
// ---------------------------------------------------------------------------
__global__ void attention_kernel(const float* __restrict__ enc_part,  // [400*32,128]
                                 const float* __restrict__ dec_part,  // [1600,128]
                                 const float* __restrict__ v_att,     // [128]
                                 const float* __restrict__ output_enc,// [400,32,1024]
                                 bf16* __restrict__ pv) {             // stride 1536
  __shared__ float decL[128];
  __shared__ float e[400];
  __shared__ float red[256];
  const int t = blockIdx.x >> 5;     // /32
  const int b = blockIdx.x & 31;
  const int tid = threadIdx.x;

  if (tid < 128) decL[tid] = dec_part[(long)(t * 32 + b) * 128 + tid];
  __syncthreads();

  for (int s = tid; s < 400; s += 256) {
    const float* ep = enc_part + (long)(s * 32 + b) * 128;
    float acc = 0.0f;
    for (int a = 0; a < 128; ++a) acc += v_att[a] * tanhf(ep[a] + decL[a]);
    e[s] = acc;
  }
  __syncthreads();

  // softmax over s (400)
  float mx = -1e30f;
  for (int s = tid; s < 400; s += 256) mx = fmaxf(mx, e[s]);
  red[tid] = mx;
  __syncthreads();
  for (int off = 128; off > 0; off >>= 1) {
    if (tid < off) red[tid] = fmaxf(red[tid], red[tid + off]);
    __syncthreads();
  }
  mx = red[0];
  __syncthreads();
  float sm = 0.0f;
  for (int s = tid; s < 400; s += 256) {
    float ex = expf(e[s] - mx);
    e[s] = ex;
    sm += ex;
  }
  red[tid] = sm;
  __syncthreads();
  for (int off = 128; off > 0; off >>= 1) {
    if (tid < off) red[tid] += red[tid + off];
    __syncthreads();
  }
  const float inv = 1.0f / red[0];
  __syncthreads();

  // context over encoder dim, written as bf16 straight into pv_in[:,512:1536]
  for (int ch = tid; ch < 1024; ch += 256) {
    float acc = 0.0f;
    for (int s = 0; s < 400; ++s)
      acc += e[s] * output_enc[(long)(s * 32 + b) * 1024 + ch];
    pv[(long)(t * 32 + b) * 1536 + 512 + ch] = (bf16)(acc * inv);
  }
}

// Copy final LSTM states into the output tail: [hidden_dec(2,32,512), cell(2,32,512)]
__global__ void tail_copy_kernel(const float* __restrict__ h0, const float* __restrict__ h1,
                                 const float* __restrict__ c0, const float* __restrict__ c1,
                                 float* __restrict__ out) {
  int id = blockIdx.x * 256 + threadIdx.x;   // 0..65535
  int which = id >> 14;
  int r = id & 16383;
  const float* s = (which == 0) ? h0 : (which == 1) ? h1 : (which == 2) ? c0 : c1;
  out[id] = s[r];
}

// ---------------------------------------------------------------------------
// Host side
// ---------------------------------------------------------------------------
extern "C" void kernel_launch(void* const* d_in, const int* in_sizes, int n_in,
                              void* d_out, int out_size, void* d_ws, size_t ws_size,
                              hipStream_t stream) {
  const float* output_enc = (const float*)d_in[0];
  const float* input_dec  = (const float*)d_in[1];
  const float* hidden_enc = (const float*)d_in[2];
  const float* cell_enc   = (const float*)d_in[3];
  const float* W_ih0 = (const float*)d_in[4];
  const float* W_hh0 = (const float*)d_in[5];
  const float* b_ih0 = (const float*)d_in[6];
  const float* b_hh0 = (const float*)d_in[7];
  const float* W_ih1 = (const float*)d_in[8];
  const float* W_hh1 = (const float*)d_in[9];
  const float* b_ih1 = (const float*)d_in[10];
  const float* b_hh1 = (const float*)d_in[11];
  const float* W_red = (const float*)d_in[12];
  const float* b_red = (const float*)d_in[13];
  const float* W_att = (const float*)d_in[14];
  const float* b_att = (const float*)d_in[15];
  const float* v_att = (const float*)d_in[16];
  const float* W_v1  = (const float*)d_in[17];
  const float* b_v1  = (const float*)d_in[18];
  const float* W_v2  = (const float*)d_in[19];
  const float* b_v2  = (const float*)d_in[20];

  // ---- workspace carve (256B aligned) ----
  char* p = (char*)d_ws;
  auto carve = [&](size_t bytes) -> char* {
    char* r = p;
    p += (bytes + 255) & ~(size_t)255;
    return r;
  };
  bf16* W_ih0_bf = (bf16*)carve((size_t)2048 * 512 * 2);
  bf16* W_hh0_bf = (bf16*)carve((size_t)2048 * 512 * 2);
  bf16* W_ih1_bf = (bf16*)carve((size_t)2048 * 512 * 2);
  bf16* W_hh1_bf = (bf16*)carve((size_t)2048 * 512 * 2);
  bf16* W_att_bf = (bf16*)carve((size_t)128 * 1536 * 2);
  bf16* W_v1_bf  = (bf16*)carve((size_t)1024 * 1536 * 2);
  bf16* W_v2_bf  = (bf16*)carve((size_t)32000 * 1024 * 2);
  bf16* indec_bf = (bf16*)carve((size_t)50 * 32 * 512 * 2);
  bf16* outenc_bf= (bf16*)carve((size_t)400 * 32 * 1024 * 2);
  float* b0c = (float*)carve(2048 * 4);
  float* b1c = (float*)carve(2048 * 4);
  float* h0  = (float*)carve(32 * 512 * 4);
  float* h1  = (float*)carve(32 * 512 * 4);
  float* c0  = (float*)carve(32 * 512 * 4);
  float* c1  = (float*)carve(32 * 512 * 4);
  bf16*  h0b = (bf16*)carve(32 * 512 * 2);
  bf16*  h1b = (bf16*)carve(32 * 512 * 2);
  float* g   = (float*)carve(32 * 2048 * 4);
  bf16*  pv  = (bf16*)carve((size_t)1600 * 1536 * 2);     // [out_dec | context]
  float* encp= (float*)carve((size_t)12800 * 128 * 4);
  float* decp= (float*)carve((size_t)1600 * 128 * 4);
  float* hid = (float*)carve((size_t)1600 * 1024 * 4);
  bf16*  hidb= (bf16*)carve((size_t)1600 * 1024 * 2);
  (void)ws_size; (void)in_sizes; (void)n_in; (void)out_size;

  auto convert = [&](const float* s, bf16* d, int n) {
    convert_f32_bf16_kernel<<<(n + 255) / 256, 256, 0, stream>>>(s, d, n);
  };
  auto gemm = [&](const bf16* A1, int lda1, const bf16* A2, int lda2,
                  const bf16* B1, int ldb1, const bf16* B2, int ldb2,
                  int Ksplit, int K, const float* bias,
                  float* C, int ldc, bf16* Cbf, int ldcbf, int M, int N) {
    dim3 blk(32, 4), grd(N / 128, M / 32);
    gemm_bf16_wmma_kernel<<<grd, blk, 0, stream>>>(
        A1, lda1, A2, lda2, B1, ldb1, B2, ldb2, Ksplit, K, bias, C, ldc, Cbf, ldcbf);
  };

  // ---- 1. bf16 conversions + combined biases ----
  convert(W_ih0, W_ih0_bf, 2048 * 512);
  convert(W_hh0, W_hh0_bf, 2048 * 512);
  convert(W_ih1, W_ih1_bf, 2048 * 512);
  convert(W_hh1, W_hh1_bf, 2048 * 512);
  convert(W_att, W_att_bf, 128 * 1536);
  convert(W_v1,  W_v1_bf,  1024 * 1536);
  convert(W_v2,  W_v2_bf,  32000 * 1024);
  convert(input_dec,  indec_bf,  50 * 32 * 512);
  convert(output_enc, outenc_bf, 400 * 32 * 1024);
  addvec_kernel<<<8, 256, 0, stream>>>(b_ih0, b_hh0, b0c, 2048);
  addvec_kernel<<<8, 256, 0, stream>>>(b_ih1, b_hh1, b1c, 2048);

  // ---- 2. reduce_dim: initial decoder h/c states ----
  reduce_dim_kernel<<<256, 256, 0, stream>>>(hidden_enc, cell_enc, W_red, b_red,
                                             h0, h1, c0, c1);
  convert(h0, h0b, 32 * 512);
  convert(h1, h1b, 32 * 512);

  // ---- 3. LSTM scan: 50 timesteps x 2 layers ----
  for (int t = 0; t < kSD; ++t) {
    // layer 0: g = [x_t | h0] @ [W_ih0 | W_hh0]^T + (b_ih0+b_hh0)
    gemm(indec_bf + (long)t * 32 * 512, 512, h0b, 512,
         W_ih0_bf, 512, W_hh0_bf, 512, 512, 1024, b0c,
         g, 2048, nullptr, 0, 32, 2048);
    lstm_cell_kernel<<<64, 256, 0, stream>>>(g, c0, h0, h0b, nullptr, 0);
    // layer 1: g = [h0 | h1] @ [W_ih1 | W_hh1]^T + (b_ih1+b_hh1)
    gemm(h0b, 512, h1b, 512,
         W_ih1_bf, 512, W_hh1_bf, 512, 512, 1024, b1c,
         g, 2048, nullptr, 0, 32, 2048);
    lstm_cell_kernel<<<64, 256, 0, stream>>>(g, c1, h1, h1b,
                                             pv + (long)t * 32 * 1536, 1536);
  }

  // ---- 4. attention projections ----
  // enc_part[12800,128] = output_enc @ W_att[:, :1024]^T
  gemm(outenc_bf, 1024, outenc_bf, 1024,
       W_att_bf, 1536, W_att_bf, 1536, 1024, 1024, nullptr,
       encp, 128, nullptr, 0, 12800, 128);
  // dec_part[1600,128] = output_dec @ W_att[:, 1024:]^T + b_att
  gemm(pv, 1536, pv, 1536,
       W_att_bf + 1024, 1536, W_att_bf + 1024, 1536, 512, 512, b_att,
       decp, 128, nullptr, 0, 1600, 128);

  // ---- 5. fused tanh / softmax / context ----
  attention_kernel<<<1600, 256, 0, stream>>>(encp, decp, v_att, output_enc, pv);

  // ---- 6. vocab MLP ----
  gemm(pv, 1536, pv, 1536, W_v1_bf, 1536, W_v1_bf, 1536, 1536, 1536, b_v1,
       hid, 1024, hidb, 1024, 1600, 1024);
  gemm(hidb, 1024, hidb, 1024, W_v2_bf, 1024, W_v2_bf, 1024, 1024, 1024, b_v2,
       (float*)d_out, 32000, nullptr, 0, 1600, 32000);

  // ---- 7. final hidden/cell states ----
  tail_copy_kernel<<<256, 256, 0, stream>>>(h0, h1, c0, c1,
                                            (float*)d_out + kPvocabElems);
}
